// AttentionFlowLSTMEncoder_64261300683117
// MI455X (gfx1250) — matrix operand order repaired and checked by reading, not compiled, not run on hardware
//
#include <hip/hip_runtime.h>
#include <stdint.h>
#include <stddef.h>

// ---------------- model constants ----------------
#define NBATCH 32
#define HIDDEN 256
#define GATES  1024   // 4*HIDDEN
#define D2     512    // 2*HIDDEN
#define PLEN   512
#define ALEN   128
#define EDIM   300
#define EPAD   320    // EDIM padded to multiple of 32 for WMMA K

typedef __bf16 bf16_t;
typedef __attribute__((ext_vector_type(8)))  bf16_t bf16x8;
typedef __attribute__((ext_vector_type(16))) bf16_t bf16x16;
typedef __attribute__((ext_vector_type(8)))  float  f32x8;

// ---------------- device helpers ----------------
__device__ __forceinline__ unsigned short f2bf_bits(float f) {
  unsigned u = __float_as_uint(f);
  unsigned rb = (u >> 16) & 1u;
  u += 0x7fffu + rb;             // round-to-nearest-even
  return (unsigned short)(u >> 16);
}
__device__ __forceinline__ float sigf(float x) {
  return 1.0f / (1.0f + __expf(-x));
}

__device__ __forceinline__ f32x8 wmma_bf16(bf16x16 a, bf16x16 b, f32x8 c) {
  // (neg_a, A, neg_b, B, c_mod, C, reuse_a, reuse_b)
  return __builtin_amdgcn_wmma_f32_16x16x32_bf16(false, a, false, b, (short)0, c,
                                                 false, false);
}

// A fragment: 16x32 bf16 (M x K), row-major source, leading dim ld.
// lane 0-15: M=lane, K=[0..8) in v0..3, [16..24) in v4..7
// lane 16-31: M=lane-16, K=[8..16) in v0..3, [24..32) in v4..7
__device__ __forceinline__ bf16x16 load_frag_a(const unsigned short* base, int ld,
                                               int row0, int k0, int lane) {
  int m  = row0 + (lane & 15);
  int kb = k0 + ((lane >> 4) << 3);
  const unsigned short* p = base + (size_t)m * ld + kb;
  union { bf16x16 v; bf16x8 h[2]; } u;
  u.h[0] = *reinterpret_cast<const bf16x8*>(p);
  u.h[1] = *reinterpret_cast<const bf16x8*>(p + 16);
  return u.v;
}

// B fragment: 32x16 (K x N); source stored N x K row-major (weight layout).
// lane 0-15: N=lane, K=[0..8),[16..24) ; lane 16-31: N=lane-16, K=[8..16),[24..32)
__device__ __forceinline__ bf16x16 load_frag_b(const unsigned short* base, int ld,
                                               int col0, int k0, int lane) {
  int n  = col0 + (lane & 15);
  int kb = k0 + ((lane >> 4) << 3);
  const unsigned short* p = base + (size_t)n * ld + kb;
  union { bf16x16 v; bf16x8 h[2]; } u;
  u.h[0] = *reinterpret_cast<const bf16x8*>(p);
  u.h[1] = *reinterpret_cast<const bf16x8*>(p + 16);
  return u.v;
}

// ---------------- elementwise kernels (division-free indexing) ----------------
__global__ void k_f32_to_bf16(const float* __restrict__ src,
                              unsigned short* __restrict__ dst, size_t n) {
  size_t i = (size_t)blockIdx.x * blockDim.x + threadIdx.x;
  if (i < n) dst[i] = f2bf_bits(src[i]);
}

// [rows][K] f32 -> [rows][Kp] bf16, zero pad.  grid: (ceil(Kp/256), rows)
__global__ void k_f32_to_bf16_pad(const float* __restrict__ src,
                                  unsigned short* __restrict__ dst, int K, int Kp) {
  int c = blockIdx.x * blockDim.x + threadIdx.x;
  size_t r = blockIdx.y;
  if (c >= Kp) return;
  dst[r * (size_t)Kp + c] = (c < K) ? f2bf_bits(src[r * (size_t)K + c]) : (unsigned short)0;
}

__global__ void k_bias_fuse(const float* __restrict__ a, const float* __restrict__ b,
                            float* __restrict__ o, int n) {
  int i = blockIdx.x * blockDim.x + threadIdx.x;
  if (i < n) o[i] = a[i] + b[i];
}

// ---------------- WMMA GEMM: C[M,N] = A[M,K](bf16) * B[N,K](bf16)^T + bias ----------------
// Fully aligned: M%128==0, N%64==0, K%32==0 (callers guarantee via padding).
// Block = 256 threads = 8 waves; block tile 128x64; wave tile 32x32 (4 wmma / K-step).
__global__ __launch_bounds__(256) void k_wmma_gemm(
    const unsigned short* __restrict__ A, const unsigned short* __restrict__ B,
    const float* __restrict__ bias, float* __restrict__ C,
    int M, int N, int K,
    long long strideA, long long strideB, long long strideC) {
  __shared__ unsigned short sA[128 * 32];  // 8 KB
  __shared__ unsigned short sB[64 * 32];   // 4 KB

  const int tid  = threadIdx.x;
  const int lane = tid & 31;
  const int wv   = tid >> 5;     // 8 waves
  const int wm   = wv & 3;       // m strip 0..3 (32 rows each)
  const int wn   = wv >> 2;      // n strip 0..1 (32 cols each)
  const int bm   = blockIdx.y * 128;
  const int bn   = blockIdx.x * 64;
  const int bat  = blockIdx.z;

  const unsigned short* Ab = A + (size_t)bat * strideA;
  const unsigned short* Bb = B + (size_t)bat * strideB;
  float* Cb = C + (size_t)bat * strideC;

  f32x8 acc[2][2] = {};

  const int arow0 = tid >> 2;          // 0..63
  const int akc   = (tid & 3) << 3;    // 0,8,16,24

  for (int k0 = 0; k0 < K; k0 += 32) {
    // stage A (128x32) and B (64x32), unconditional b128 loads
    *reinterpret_cast<bf16x8*>(&sA[arow0 * 32 + akc]) =
        *reinterpret_cast<const bf16x8*>(Ab + (size_t)(bm + arow0) * K + k0 + akc);
    *reinterpret_cast<bf16x8*>(&sA[(arow0 + 64) * 32 + akc]) =
        *reinterpret_cast<const bf16x8*>(Ab + (size_t)(bm + arow0 + 64) * K + k0 + akc);
    *reinterpret_cast<bf16x8*>(&sB[arow0 * 32 + akc]) =
        *reinterpret_cast<const bf16x8*>(Bb + (size_t)(bn + arow0) * K + k0 + akc);
    if (k0 + 32 < K) {  // speculative prefetch of next K tile -> global_prefetch_b8
      __builtin_prefetch(Ab + (size_t)(bm + arow0) * K + k0 + 32 + akc, 0, 1);
      __builtin_prefetch(Bb + (size_t)(bn + arow0) * K + k0 + 32 + akc, 0, 1);
    }
    __syncthreads();

    bf16x16 a0 = load_frag_a(sA, 32, wm * 32 + 0, 0, lane);
    bf16x16 a1 = load_frag_a(sA, 32, wm * 32 + 16, 0, lane);
    bf16x16 b0 = load_frag_b(sB, 32, wn * 32 + 0, 0, lane);
    bf16x16 b1 = load_frag_b(sB, 32, wn * 32 + 16, 0, lane);
    acc[0][0] = wmma_bf16(a0, b0, acc[0][0]);
    acc[0][1] = wmma_bf16(a0, b1, acc[0][1]);
    acc[1][0] = wmma_bf16(a1, b0, acc[1][0]);
    acc[1][1] = wmma_bf16(a1, b1, acc[1][1]);
    __syncthreads();
  }

  // C layout: lane L -> col = L&15 ; VGPR v -> row = v + 8*(L>>4)
  const int colBase = bn + wn * 32 + (lane & 15);
  const int rowBase = bm + wm * 32 + ((lane >> 4) << 3);
  for (int mt = 0; mt < 2; ++mt)
    for (int nt = 0; nt < 2; ++nt) {
      int col = colBase + nt * 16;
      float bv = bias ? bias[col] : 0.0f;
      for (int v = 0; v < 8; ++v) {
        int row = rowBase + mt * 16 + v;
        Cb[(size_t)row * N + col] = acc[mt][nt][v] + bv;
      }
    }
}

// ---------------- persistent LSTM sequence kernel ----------------
// One workgroup per direction (blockIdx.x: 0 fwd, 1 bwd), 1024 threads = 32 waves.
// Per step: gates(32x1024) = h(32x256,bf16 in LDS) @ Whh^T (bf16, L2-resident) + gx,
// then elementwise cell update; c in registers, h re-published to LDS as bf16.
// Dynamic LDS: 32*256 bf16 + 32*1024 f32 = 147456 B (< 320KB WGP LDS).
__global__ __launch_bounds__(1024) void k_lstm_seq(
    const float* __restrict__ gx_f, const float* __restrict__ gx_b,
    const unsigned short* __restrict__ Whh_f, const unsigned short* __restrict__ Whh_b,
    float* __restrict__ hout_f, float* __restrict__ hout_b, int T) {
  extern __shared__ char smem[];
  unsigned short* hbf = reinterpret_cast<unsigned short*>(smem);                   // [32][256]
  float* gbuf = reinterpret_cast<float*>(smem + NBATCH * HIDDEN * sizeof(unsigned short)); // [32][1024]

  const int dir = blockIdx.x;
  const float* gx = dir ? gx_b : gx_f;
  const unsigned short* Whh = dir ? Whh_b : Whh_f;
  float* hout = dir ? hout_b : hout_f;

  const int tid   = threadIdx.x;
  const int lane  = tid & 31;
  const int wv    = tid >> 5;        // 0..31
  const int nbase = wv * 32;         // gate-column strip owned by this wave

  const int cb  = tid >> 5;          // batch owned for cell update
  const int cj0 = (tid & 31) * 8;    // hidden j range [cj0, cj0+8)
  float creg[8];
  for (int u = 0; u < 8; ++u) creg[u] = 0.0f;

  for (int i = tid; i < NBATCH * HIDDEN / 2; i += 1024)
    reinterpret_cast<unsigned*>(hbf)[i] = 0u;
  __syncthreads();

  for (int s = 0; s < T; ++s) {
    const int t = dir ? (T - 1 - s) : s;

    // ---- recurrent GEMM: (32 x 256) x (256 x 32-strip) via WMMA ----
    f32x8 acc[2][2] = {};
#pragma unroll
    for (int k0 = 0; k0 < HIDDEN; k0 += 32) {
      bf16x16 a0 = load_frag_a(hbf, HIDDEN, 0, k0, lane);
      bf16x16 a1 = load_frag_a(hbf, HIDDEN, 16, k0, lane);
      bf16x16 b0 = load_frag_b(Whh, HIDDEN, nbase + 0, k0, lane);
      bf16x16 b1 = load_frag_b(Whh, HIDDEN, nbase + 16, k0, lane);
      acc[0][0] = wmma_bf16(a0, b0, acc[0][0]);
      acc[0][1] = wmma_bf16(a0, b1, acc[0][1]);
      acc[1][0] = wmma_bf16(a1, b0, acc[1][0]);
      acc[1][1] = wmma_bf16(a1, b1, acc[1][1]);
    }
    const int colBase = nbase + (lane & 15);
    const int rowBase = (lane >> 4) << 3;
    for (int mt = 0; mt < 2; ++mt)
      for (int nt = 0; nt < 2; ++nt) {
        int gidx = colBase + nt * 16;
        for (int v = 0; v < 8; ++v) {
          int b = mt * 16 + rowBase + v;
          gbuf[b * GATES + gidx] =
              acc[mt][nt][v] + gx[((size_t)b * T + t) * GATES + gidx];
        }
      }
    __syncthreads();

    // ---- elementwise cell update ----
    for (int u = 0; u < 8; ++u) {
      int j = cj0 + u;
      float gi = gbuf[cb * GATES + j];
      float gf = gbuf[cb * GATES + HIDDEN + j];
      float gg = gbuf[cb * GATES + 2 * HIDDEN + j];
      float go = gbuf[cb * GATES + 3 * HIDDEN + j];
      float c = sigf(gf) * creg[u] + sigf(gi) * tanhf(gg);
      creg[u] = c;
      float h = sigf(go) * tanhf(c);
      hout[((size_t)cb * T + t) * HIDDEN + j] = h;
      hbf[cb * HIDDEN + j] = f2bf_bits(h);
    }
    __syncthreads();
  }
}

// concat fwd/bwd hidden -> [NT][2H].  grid: (H/256, NT)
__global__ void k_concat2(const float* __restrict__ a, const float* __restrict__ b,
                          float* __restrict__ of, unsigned short* __restrict__ ob, int H) {
  int j = blockIdx.x * blockDim.x + threadIdx.x;
  size_t nt = blockIdx.y;
  if (j >= H) return;
  size_t i = nt * (size_t)H + j;
  float va = a[i], vb = b[i];
  size_t o0 = nt * (size_t)(2 * H) + j;
  of[o0] = va;
  of[o0 + H] = vb;
  if (ob) { ob[o0] = f2bf_bits(va); ob[o0 + H] = f2bf_bits(vb); }
}

// ---------------- attention kernels ----------------
__global__ void k_att_prepH(const float* __restrict__ H, const float* __restrict__ w,
                            unsigned short* __restrict__ Hw, float* __restrict__ Hwh) {
  __shared__ float red[256];
  int r = blockIdx.x, tid = threadIdx.x;
  const float* wh = w;
  const float* we = w + 2 * D2;
  float s = 0.0f;
  for (int d = tid; d < D2; d += 256) {
    float v = H[(size_t)r * D2 + d];
    s += v * wh[d];
    Hw[(size_t)r * D2 + d] = f2bf_bits(v * we[d]);
  }
  red[tid] = s;
  __syncthreads();
  for (int k = 128; k > 0; k >>= 1) {
    if (tid < k) red[tid] += red[tid + k];
    __syncthreads();
  }
  if (tid == 0) Hwh[r] = red[0];
}

__global__ void k_att_prepU(const float* __restrict__ U, const float* __restrict__ w,
                            unsigned short* __restrict__ Ub, float* __restrict__ Uwu) {
  __shared__ float red[256];
  int r = blockIdx.x, tid = threadIdx.x;
  const float* wu = w + D2;
  float s = 0.0f;
  for (int d = tid; d < D2; d += 256) {
    float v = U[(size_t)r * D2 + d];
    s += v * wu[d];
    Ub[(size_t)r * D2 + d] = f2bf_bits(v);
  }
  red[tid] = s;
  __syncthreads();
  for (int k = 128; k > 0; k >>= 1) {
    if (tid < k) red[tid] += red[tid + k];
    __syncthreads();
  }
  if (tid == 0) Uwu[r] = red[0];
}

// softmax over A of (sim_core + Hwh + Uwu); emit row max (b).  grid: (ceil(P/256), N)
__global__ void k_softmax_att(const float* __restrict__ sim, const float* __restrict__ Hwh,
                              const float* __restrict__ Uwu, unsigned short* __restrict__ abf,
                              float* __restrict__ bmax, int P, int A) {
  int p = blockIdx.x * blockDim.x + threadIdx.x;
  int n = blockIdx.y;
  if (p >= P) return;
  size_t r = (size_t)n * P + p;
  const float* srow = sim + r * A;
  const float* uw = Uwu + (size_t)n * A;
  float ht = Hwh[r];
  float mx = -1e30f;
  for (int a = 0; a < A; ++a) mx = fmaxf(mx, srow[a] + ht + uw[a]);
  bmax[r] = mx;
  float ssum = 0.0f;
  for (int a = 0; a < A; ++a) ssum += __expf(srow[a] + ht + uw[a] - mx);
  float inv = 1.0f / ssum;
  for (int a = 0; a < A; ++a)
    abf[r * A + a] = f2bf_bits(__expf(srow[a] + ht + uw[a] - mx) * inv);
}

// U[n][A][d2] -> UT[n][d2][A] (bf16).  grid: (d2/256, A, NB)
__global__ void k_transpose_bf16(const float* __restrict__ U, unsigned short* __restrict__ UT,
                                 int A, int d2) {
  int d = blockIdx.x * blockDim.x + threadIdx.x;
  int a = blockIdx.y;
  int n = blockIdx.z;
  if (d >= d2) return;
  UT[((size_t)n * d2 + d) * A + a] =
      f2bf_bits(U[((size_t)n * A + a) * d2 + d]);
}

// Ht[n,d] = sum_p bmax[n,p] * H[n,p,d].  grid: (d2/256, NB)
__global__ void k_weighted_H(const float* __restrict__ bmax, const float* __restrict__ H,
                             float* __restrict__ Ht, int P, int d2) {
  int n = blockIdx.y;
  int d = blockIdx.x * blockDim.x + threadIdx.x;
  if (d >= d2) return;
  float s = 0.0f;
  for (int p = 0; p < P; ++p)
    s += bmax[(size_t)n * P + p] * H[(((size_t)n * P) + p) * d2 + d];
  Ht[(size_t)n * d2 + d] = s;
}

// G[n,p,:] = [H, Ut, H*Ut, H*Ht[n]] (bf16, 4*d2 wide).  grid: (d2/256, P, NB)
__global__ void k_build_G(const float* __restrict__ H, const float* __restrict__ Ut,
                          const float* __restrict__ Ht, unsigned short* __restrict__ G,
                          int P, int d2) {
  int d = blockIdx.x * blockDim.x + threadIdx.x;
  int p = blockIdx.y;
  int n = blockIdx.z;
  if (d >= d2) return;
  size_t np = (size_t)n * P + p;
  size_t i = np * d2 + d;
  float h = H[i];
  float ut = Ut[i];
  float ht = Ht[(size_t)n * d2 + d];
  size_t base = np * (size_t)(4 * d2) + d;
  G[base]          = f2bf_bits(h);
  G[base + d2]     = f2bf_bits(ut);
  G[base + 2 * d2] = f2bf_bits(h * ut);
  G[base + 3 * d2] = f2bf_bits(h * ht);
}

// mean over time: x[N][T][D] -> o[N][D].  grid: (D/256, NB)
__global__ void k_mean_T(const float* __restrict__ x, float* __restrict__ o, int T, int D) {
  int n = blockIdx.y;
  int d = blockIdx.x * blockDim.x + threadIdx.x;
  if (d >= D) return;
  float s = 0.0f;
  for (int t = 0; t < T; ++t) s += x[(((size_t)n * T) + t) * D + d];
  o[(size_t)n * D + d] = s / (float)T;
}

// out[n,e] = mean[n,:] . W[e,:] + b[e].  grid: (ceil(E/256), NB)
__global__ void k_fc(const float* __restrict__ x, const float* __restrict__ W,
                     const float* __restrict__ b, float* __restrict__ o, int E, int D) {
  int n = blockIdx.y;
  int e = blockIdx.x * blockDim.x + threadIdx.x;
  if (e >= E) return;
  float s = b[e];
  for (int d = 0; d < D; ++d) s += x[(size_t)n * D + d] * W[(size_t)e * D + d];
  o[(size_t)n * E + e] = s;
}

// ---------------- host orchestration ----------------
static inline int ceildiv_i(long long a, long long b) { return (int)((a + b - 1) / b); }

static void launch_gemm(hipStream_t s, const unsigned short* A, const unsigned short* B,
                        const float* bias, float* C, int M, int N, int K,
                        long long sA, long long sB, long long sC, int batch) {
  dim3 g(N / 64, M / 128, batch);   // callers guarantee M%128==0, N%64==0, K%32==0
  k_wmma_gemm<<<g, 256, 0, s>>>(A, B, bias, C, M, N, K, sA, sB, sC);
}

struct LstmScratch {
  unsigned short *wihF, *whhF, *wihB, *whhB;
  float *biasF, *biasB, *gxF, *gxB, *houtF, *houtB;
};

static const size_t LSTM_SMEM =
    (size_t)NBATCH * HIDDEN * sizeof(unsigned short) + (size_t)NBATCH * GATES * sizeof(float);

// xb: [NB*T][Dp] bf16 (zero-padded to Dp); runs one bidirectional layer.
static void run_bilstm_layer(hipStream_t s, const unsigned short* xb, int T, int D, int Dp,
                             const float* WihF, const float* WhhF, const float* bihF,
                             const float* bhhF, const float* WihB, const float* WhhB,
                             const float* bihB, const float* bhhB, const LstmScratch& sc,
                             float* out_f32, unsigned short* out_bf16) {
  const int NT = NBATCH * T;
  size_t nWhh = (size_t)GATES * HIDDEN;
  dim3 gW(ceildiv_i(Dp, 256), GATES);
  k_f32_to_bf16_pad<<<gW, 256, 0, s>>>(WihF, sc.wihF, D, Dp);
  k_f32_to_bf16_pad<<<gW, 256, 0, s>>>(WihB, sc.wihB, D, Dp);
  k_f32_to_bf16<<<ceildiv_i((long long)nWhh, 256), 256, 0, s>>>(WhhF, sc.whhF, nWhh);
  k_f32_to_bf16<<<ceildiv_i((long long)nWhh, 256), 256, 0, s>>>(WhhB, sc.whhB, nWhh);
  k_bias_fuse<<<4, 256, 0, s>>>(bihF, bhhF, sc.biasF, GATES);
  k_bias_fuse<<<4, 256, 0, s>>>(bihB, bhhB, sc.biasB, GATES);
  launch_gemm(s, xb, sc.wihF, sc.biasF, sc.gxF, NT, GATES, Dp, 0, 0, 0, 1);
  launch_gemm(s, xb, sc.wihB, sc.biasB, sc.gxB, NT, GATES, Dp, 0, 0, 0, 1);
  k_lstm_seq<<<dim3(2), dim3(1024), LSTM_SMEM, s>>>(sc.gxF, sc.gxB, sc.whhF, sc.whhB,
                                                    sc.houtF, sc.houtB, T);
  dim3 gC(HIDDEN / 256, NT);
  k_concat2<<<gC, 256, 0, s>>>(sc.houtF, sc.houtB, out_f32, out_bf16, HIDDEN);
}

extern "C" void kernel_launch(void* const* d_in, const int* in_sizes, int n_in,
                              void* d_out, int out_size, void* d_ws, size_t ws_size,
                              hipStream_t stream) {
  (void)in_sizes; (void)n_in; (void)out_size; (void)ws_size;

  const float* passage = (const float*)d_in[0];
  const float* answer  = (const float*)d_in[1];
  // params flattened depth-first: pass_lstm(2 layers x {fwd,bwd} x {Wih,Whh,bih,bhh}),
  // ans_lstm(same), enc_lstm(1 layer), att_w, fc_W, fc_b
  auto PW = [&](int base, int l, int dir, int which) -> const float* {
    return (const float*)d_in[base + (l * 2 + dir) * 4 + which];
  };
  const int PASS_B = 2, ANS_B = 18, ENC_B = 34;
  const float* att_w = (const float*)d_in[42];
  const float* fc_W  = (const float*)d_in[43];
  const float* fc_b  = (const float*)d_in[44];
  float* out = (float*)d_out;

  // bump allocator over workspace
  char* ws = (char*)d_ws;
  size_t cur = 0;
  auto alloc = [&](size_t bytes) -> void* {
    void* p = ws + cur;
    cur += (bytes + 255) & ~(size_t)255;
    return p;
  };

  const size_t NTP = (size_t)NBATCH * PLEN;   // 16384
  const size_t NTA = (size_t)NBATCH * ALEN;   // 4096

  LstmScratch sc;
  sc.wihF  = (unsigned short*)alloc((size_t)GATES * 2048 * 2);
  sc.wihB  = (unsigned short*)alloc((size_t)GATES * 2048 * 2);
  sc.whhF  = (unsigned short*)alloc((size_t)GATES * HIDDEN * 2);
  sc.whhB  = (unsigned short*)alloc((size_t)GATES * HIDDEN * 2);
  sc.biasF = (float*)alloc(GATES * 4);
  sc.biasB = (float*)alloc(GATES * 4);
  sc.gxF   = (float*)alloc(NTP * GATES * 4);
  sc.gxB   = (float*)alloc(NTP * GATES * 4);
  sc.houtF = (float*)alloc(NTP * HIDDEN * 4);
  sc.houtB = (float*)alloc(NTP * HIDDEN * 4);

  unsigned short* xb  = (unsigned short*)alloc(NTP * 2048 * 2);  // layer inputs / G (bf16)
  float* xA   = (float*)alloc(NTP * D2 * 4);   // ping layer output / enc output
  float* Hbuf = (float*)alloc(NTP * D2 * 4);   // passage encoding
  float* Ubuf = (float*)alloc(NTA * D2 * 4);   // answer encoding
  unsigned short* Hw  = (unsigned short*)alloc(NTP * D2 * 2);
  unsigned short* Ub  = (unsigned short*)alloc(NTA * D2 * 2);
  float* Hwh  = (float*)alloc(NTP * 4);
  float* Uwu  = (float*)alloc(NTA * 4);
  float* simc = (float*)alloc((size_t)NBATCH * PLEN * ALEN * 4);
  unsigned short* abf = (unsigned short*)alloc((size_t)NBATCH * PLEN * ALEN * 2);
  unsigned short* UTb = (unsigned short*)alloc((size_t)NBATCH * D2 * ALEN * 2);
  float* bmax = (float*)alloc(NTP * 4);
  float* Ut   = (float*)alloc(NTP * D2 * 4);
  float* Ht   = (float*)alloc((size_t)NBATCH * D2 * 4);
  float* meanb = (float*)alloc((size_t)NBATCH * D2 * 4);

  // ---- passage BiLSTM (2 layers) ----
  k_f32_to_bf16_pad<<<dim3(ceildiv_i(EPAD, 256), (unsigned)NTP), 256, 0, stream>>>(
      passage, xb, EDIM, EPAD);
  run_bilstm_layer(stream, xb, PLEN, EDIM, EPAD,
                   PW(PASS_B,0,0,0), PW(PASS_B,0,0,1), PW(PASS_B,0,0,2), PW(PASS_B,0,0,3),
                   PW(PASS_B,0,1,0), PW(PASS_B,0,1,1), PW(PASS_B,0,1,2), PW(PASS_B,0,1,3),
                   sc, xA, xb);
  run_bilstm_layer(stream, xb, PLEN, D2, D2,
                   PW(PASS_B,1,0,0), PW(PASS_B,1,0,1), PW(PASS_B,1,0,2), PW(PASS_B,1,0,3),
                   PW(PASS_B,1,1,0), PW(PASS_B,1,1,1), PW(PASS_B,1,1,2), PW(PASS_B,1,1,3),
                   sc, Hbuf, xb);

  // ---- answer BiLSTM (2 layers) ----
  k_f32_to_bf16_pad<<<dim3(ceildiv_i(EPAD, 256), (unsigned)NTA), 256, 0, stream>>>(
      answer, xb, EDIM, EPAD);
  run_bilstm_layer(stream, xb, ALEN, EDIM, EPAD,
                   PW(ANS_B,0,0,0), PW(ANS_B,0,0,1), PW(ANS_B,0,0,2), PW(ANS_B,0,0,3),
                   PW(ANS_B,0,1,0), PW(ANS_B,0,1,1), PW(ANS_B,0,1,2), PW(ANS_B,0,1,3),
                   sc, xA, xb);
  run_bilstm_layer(stream, xb, ALEN, D2, D2,
                   PW(ANS_B,1,0,0), PW(ANS_B,1,0,1), PW(ANS_B,1,0,2), PW(ANS_B,1,0,3),
                   PW(ANS_B,1,1,0), PW(ANS_B,1,1,1), PW(ANS_B,1,1,2), PW(ANS_B,1,1,3),
                   sc, Ubuf, nullptr);

  // ---- attention flow ----
  k_att_prepH<<<(int)NTP, 256, 0, stream>>>(Hbuf, att_w, Hw, Hwh);
  k_att_prepU<<<(int)NTA, 256, 0, stream>>>(Ubuf, att_w, Ub, Uwu);
  // sim_core[n] = Hw[n] (P x d2) @ Ub[n]^T (A x d2)
  launch_gemm(stream, Hw, Ub, nullptr, simc, PLEN, ALEN, D2,
              (long long)PLEN * D2, (long long)ALEN * D2, (long long)PLEN * ALEN, NBATCH);
  k_softmax_att<<<dim3(ceildiv_i(PLEN, 256), NBATCH), 256, 0, stream>>>(simc, Hwh, Uwu, abf,
                                                                        bmax, PLEN, ALEN);
  k_transpose_bf16<<<dim3(D2 / 256, ALEN, NBATCH), 256, 0, stream>>>(Ubuf, UTb, ALEN, D2);
  // U_t[n] = a[n] (P x A) @ UT[n]^T (d2 x A)
  launch_gemm(stream, abf, UTb, nullptr, Ut, PLEN, D2, ALEN,
              (long long)PLEN * ALEN, (long long)D2 * ALEN, (long long)PLEN * D2, NBATCH);
  k_weighted_H<<<dim3(D2 / 256, NBATCH), 256, 0, stream>>>(bmax, Hbuf, Ht, PLEN, D2);
  k_build_G<<<dim3(D2 / 256, PLEN, NBATCH), 256, 0, stream>>>(Hbuf, Ut, Ht, xb, PLEN, D2);

  // ---- encoder BiLSTM (1 layer, input 8H=2048) ----
  run_bilstm_layer(stream, xb, PLEN, 4 * D2, 4 * D2,
                   PW(ENC_B,0,0,0), PW(ENC_B,0,0,1), PW(ENC_B,0,0,2), PW(ENC_B,0,0,3),
                   PW(ENC_B,0,1,0), PW(ENC_B,0,1,1), PW(ENC_B,0,1,2), PW(ENC_B,0,1,3),
                   sc, xA, nullptr);

  // ---- mean over time + FC ----
  k_mean_T<<<dim3(D2 / 256, NBATCH), 256, 0, stream>>>(xA, meanb, PLEN, D2);
  k_fc<<<dim3(ceildiv_i(EDIM, 256), NBATCH), 256, 0, stream>>>(meanb, fc_W, fc_b, out,
                                                               EDIM, D2);
}
